// StackedAttention_89283780149535
// MI455X (gfx1250) — matrix operand order repaired and checked by reading, compiled, not run on hardware
//
#include <hip/hip_runtime.h>

// ---------------------------------------------------------------------------
// CDNA5 / gfx1250 implementation of the stacked point-cloud self-attention
// reference. All heavy math (≈340 GFLOP) runs through v_wmma_f32_16x16x32_bf16
// on bf16 operands with f32 accumulation; softmax/BN epilogues stay in f32.
// GEMM tiles stream memory->LDS with GLOBAL_LOAD_ASYNC_TO_LDS_B128 (ASYNCcnt),
// double-buffered, one workgroup barrier per 32-wide K step.
// ---------------------------------------------------------------------------

typedef __bf16 bf16;
typedef __attribute__((ext_vector_type(16))) __bf16 bf16x16;
typedef __attribute__((ext_vector_type(8)))  float  f32x8;

// Direct global -> LDS async copy (16B per lane). LDS destination is the
// workgroup-relative byte offset (low 32 bits of the generic shared pointer).
__device__ __forceinline__ void async_load_b128(unsigned lds_off, const void* gptr)
{
  asm volatile("global_load_async_to_lds_b128 %0, %1, off"
               :
               : "v"(lds_off), "v"((unsigned long long)(uintptr_t)gptr)
               : "memory");
}
__device__ __forceinline__ void wait_async0()
{
  asm volatile("s_wait_asynccnt 0x0" ::: "memory");
}

// ---------------------------------------------------------------------------
// Tiled WMMA GEMM:  Cmat[m, j] = sum_k A[m*K + k] * Bt[j*K + k]  (+ bias)
// A: [M, K] row-major bf16.   Bt: [Ncols, K] row-major bf16 (i.e. B^T).
// 128x128 tile per workgroup, 256 threads = 8 waves (2 x 4), K-step = 32,
// double-buffered LDS fed by async loads. Each wave owns a 64x32 sub-tile
// = 4x2 wmma 16x16 accumulators.
// BIASMODE: 0 = none, 1 = bias[row], 2 = bias[col] (compile-time).
// WF32 / WBF: write f32 and/or bf16 outputs (compile-time).
// All M, Ncols are multiples of 128 and K of 32 (sizes are 256/2048), so no
// bounds checks are needed.
// ---------------------------------------------------------------------------
#define LDSP 40   // LDS row pitch in bf16 elements (32 + 8 pad -> 80B rows)

template <int BIASMODE, bool WF32, bool WBF>
__global__ __launch_bounds__(256) void gemm_bf16_wmma(
    const bf16* __restrict__ A, const bf16* __restrict__ Bt,
    float* __restrict__ Cout, bf16* __restrict__ CoutBf,
    const float* __restrict__ bias,
    int K, int ldc,
    long long strideA, long long strideB, long long strideC)
{
  __shared__ __align__(16) bf16 lA[2][128 * LDSP];
  __shared__ __align__(16) bf16 lB[2][128 * LDSP];

  const long long zb = blockIdx.z;
  A  += zb * strideA;
  Bt += zb * strideB;
  const long long cbase = zb * strideC;

  const int m0 = blockIdx.y * 128;
  const int n0 = blockIdx.x * 128;

  const int tid  = threadIdx.x;
  const int lane = tid & 31;
  const int wid  = tid >> 5;      // 0..7
  const int wrow = wid >> 2;      // 0..1  -> 64-row slab
  const int wcol = wid & 3;       // 0..3  -> 32-col slab
  const int half = lane >> 4;     // lane half (ISA 7.12.2 layouts)
  const int l16  = lane & 15;

  f32x8 acc[4][2];
#pragma unroll
  for (int i = 0; i < 4; ++i)
#pragma unroll
    for (int j = 0; j < 2; ++j)
#pragma unroll
      for (int e = 0; e < 8; ++e) acc[i][j][e] = 0.f;

  // global -> LDS: each thread moves 2 x 16B per operand per K-step
  const int ldRow = tid >> 2;        // 0..63
  const int ldQ   = (tid & 3) * 8;   // 0,8,16,24 (bf16 elements)

  auto issue_tile = [&](int buf, int k0) {
#pragma unroll
    for (int it = 0; it < 2; ++it) {
      const int row = ldRow + it * 64;
      async_load_b128((unsigned)(uintptr_t)&lA[buf][row * LDSP + ldQ],
                      A + (long long)(m0 + row) * K + k0 + ldQ);
      async_load_b128((unsigned)(uintptr_t)&lB[buf][row * LDSP + ldQ],
                      Bt + (long long)(n0 + row) * K + k0 + ldQ);
    }
  };

  issue_tile(0, 0);

  int buf = 0;
  for (int k0 = 0; k0 < K; k0 += 32, buf ^= 1) {
    wait_async0();        // my async writes for `buf` are done
    __syncthreads();      // ... and everyone else's too
    if (k0 + 32 < K) issue_tile(buf ^ 1, k0 + 32);

    // A fragments: lanes 0-15 hold K{0..7,16..23}, lanes 16-31 K{8..15,24..31}
    bf16x16 af[4];
#pragma unroll
    for (int mi = 0; mi < 4; ++mi) {
      const bf16* p = &lA[buf][(wrow * 64 + mi * 16 + l16) * LDSP + half * 8];
      union { bf16x16 v; uint4 q[2]; } u;
      u.q[0] = *(const uint4*)p;
      u.q[1] = *(const uint4*)(p + 16);
      af[mi] = u.v;
    }
    // B fragments: lanes 0-15 hold K{0..7,16..23}, lanes 16-31 K{8..15,24..31}
    bf16x16 bfr[2];
#pragma unroll
    for (int ni = 0; ni < 2; ++ni) {
      const bf16* p = &lB[buf][(wcol * 32 + ni * 16 + l16) * LDSP + half * 8];
      union { bf16x16 v; uint4 q[2]; } u;
      u.q[0] = *(const uint4*)p;
      u.q[1] = *(const uint4*)(p + 16);
      bfr[ni] = u.v;
    }

#pragma unroll
    for (int mi = 0; mi < 4; ++mi)
#pragma unroll
      for (int ni = 0; ni < 2; ++ni)
        acc[mi][ni] = __builtin_amdgcn_wmma_f32_16x16x32_bf16(
            false, af[mi], false, bfr[ni], (short)0, acc[mi][ni], false, false);
  }

  // Epilogue: C/D layout — VGPR r: lanes 0-15 -> M = r, lanes 16-31 -> M = 8+r
#pragma unroll
  for (int mi = 0; mi < 4; ++mi) {
    const int row0 = m0 + wrow * 64 + mi * 16 + half * 8;
    float brow[8];
    if constexpr (BIASMODE == 1) {
#pragma unroll
      for (int r = 0; r < 8; ++r) brow[r] = bias[row0 + r];
    }
#pragma unroll
    for (int ni = 0; ni < 2; ++ni) {
      const int col = n0 + wcol * 32 + ni * 16 + l16;
      float cadd = 0.f;
      if constexpr (BIASMODE == 2) cadd = bias[col];
      long long off = cbase + (long long)row0 * ldc + col;
#pragma unroll
      for (int r = 0; r < 8; ++r, off += ldc) {
        float v = acc[mi][ni][r] + cadd;
        if constexpr (BIASMODE == 1) v += brow[r];
        if constexpr (WF32) Cout[off] = v;
        if constexpr (WBF)  CoutBf[off] = (bf16)v;
      }
    }
  }
}

// ---------------------------------------------------------------------------
// f32 -> bf16 elementwise (weights). Exact-grid launches only.
// ---------------------------------------------------------------------------
__global__ __launch_bounds__(256) void cvt_f32_bf16(
    const float* __restrict__ in, bf16* __restrict__ out)
{
  const long long i = (long long)blockIdx.x * 256 + threadIdx.x;
  out[i] = (bf16)in[i];
}

// ---------------------------------------------------------------------------
// Tiled transpose: in [R, Cc] -> out [Cc, R] (f32 + bf16 shadow),
// batched over blockIdx.z with element strides inS / outS.
// ---------------------------------------------------------------------------
__global__ __launch_bounds__(256) void transpose_f32(
    const float* __restrict__ in, float* __restrict__ out, bf16* __restrict__ outbf,
    int R, int Cc, long long inS, long long outS)
{
  __shared__ float t[32][33];
  const float* ip = in + (long long)blockIdx.z * inS;
  const long long ob = (long long)blockIdx.z * outS;
  const int r0 = blockIdx.y * 32, c0 = blockIdx.x * 32;
  const int tx = threadIdx.x, ty = threadIdx.y;
#pragma unroll
  for (int j = 0; j < 32; j += 8)
    t[ty + j][tx] = ip[(long long)(r0 + ty + j) * Cc + c0 + tx];
  __syncthreads();
#pragma unroll
  for (int j = 0; j < 32; j += 8) {
    const long long o = ob + (long long)(c0 + ty + j) * R + r0 + tx;
    const float v = t[tx][ty + j];
    out[o]   = v;
    outbf[o] = (bf16)v;
  }
}

// ---------------------------------------------------------------------------
// q/k production: q and k share the GEMM (same weight), only biases differ.
// qk: [B*N, C] f32 -> q_bf16, k_bf16 (same layout).
// ---------------------------------------------------------------------------
__global__ __launch_bounds__(256) void qk_bias_cvt(
    const float* __restrict__ qk, const float* __restrict__ qb,
    const float* __restrict__ kb, bf16* __restrict__ q, bf16* __restrict__ k,
    int C)
{
  const long long idx = (long long)blockIdx.x * 256 + threadIdx.x;
  const int c = (int)(idx & (long long)(C - 1));   // C = 256 (pow2)
  const float v = qk[idx];
  q[idx] = (bf16)(v + qb[c]);
  k[idx] = (bf16)(v + kb[c]);
}

// ---------------------------------------------------------------------------
// Row softmax (axis=-1) in place on E [rows, N], N = 2048, 256 threads/row.
// ---------------------------------------------------------------------------
__global__ __launch_bounds__(256) void softmax_rows(float* __restrict__ E, int N)
{
  __shared__ float red[256];
  float* row = E + (long long)blockIdx.x * N;
  const int tid = threadIdx.x;
  float v[8];
  float m = -3.0e38f;
#pragma unroll
  for (int i = 0; i < 8; ++i) { v[i] = row[tid + i * 256]; m = fmaxf(m, v[i]); }
  red[tid] = m; __syncthreads();
  for (int s = 128; s > 0; s >>= 1) {
    if (tid < s) red[tid] = fmaxf(red[tid], red[tid + s]);
    __syncthreads();
  }
  m = red[0]; __syncthreads();
  float sum = 0.f;
#pragma unroll
  for (int i = 0; i < 8; ++i) { v[i] = __expf(v[i] - m); sum += v[i]; }
  red[tid] = sum; __syncthreads();
  for (int s = 128; s > 0; s >>= 1) {
    if (tid < s) red[tid] += red[tid + s];
    __syncthreads();
  }
  const float inv = 1.0f / red[0];
#pragma unroll
  for (int i = 0; i < 8; ++i) row[tid + i * 256] = v[i] * inv;
}

// ---------------------------------------------------------------------------
// Column sums of att [N, N] over the query axis (rows): cs[m] = sum_n E[n,m].
// ---------------------------------------------------------------------------
__global__ __launch_bounds__(256) void colsum_kernel(
    const float* __restrict__ E, float* __restrict__ cs, int N)
{
  const int col = blockIdx.x * 256 + threadIdx.x;
  float s = 0.f;
  for (int n = 0; n < N; ++n) s += E[(long long)n * N + col];
  cs[col] = s;
}

// ---------------------------------------------------------------------------
// attT[m, n] = bf16( att[n, m] / (1e-9 + cs[m]) )  — fused L1-renorm over the
// query axis, transpose, and bf16 convert so the v·att GEMM is K-contiguous.
// ---------------------------------------------------------------------------
__global__ __launch_bounds__(256) void renorm_transpose(
    const float* __restrict__ E, const float* __restrict__ cs,
    bf16* __restrict__ attT, int N)
{
  __shared__ float t[32][33];
  const int x0 = blockIdx.x * 32;   // m (column of E)
  const int y0 = blockIdx.y * 32;   // n (row of E)
  const int tx = threadIdx.x, ty = threadIdx.y;
#pragma unroll
  for (int j = 0; j < 32; j += 8)
    t[ty + j][tx] = E[(long long)(y0 + ty + j) * N + x0 + tx];
  __syncthreads();
#pragma unroll
  for (int j = 0; j < 32; j += 8) {
    const int m = x0 + ty + j;
    attT[(long long)m * N + y0 + tx] = (bf16)(t[tx][ty + j] / (1e-9f + cs[m]));
  }
}

// ---------------------------------------------------------------------------
// xmr = bf16(x - x_r), elementwise on [B*N*C].
// ---------------------------------------------------------------------------
__global__ __launch_bounds__(256) void sub_cvt(
    const float* __restrict__ a, const float* __restrict__ b, bf16* __restrict__ o)
{
  const long long i = (long long)blockIdx.x * 256 + threadIdx.x;
  o[i] = (bf16)(a[i] - b[i]);
}

// ---------------------------------------------------------------------------
// BatchNorm stats (training mode, biased variance over B*N per channel).
// y: [rows, C].  One thread per channel, each block covers rowsPer rows.
// ss layout: [0,C) sum, [C,2C) sumsq, [2C,3C) scale, [3C,4C) shift.
// ---------------------------------------------------------------------------
__global__ __launch_bounds__(256) void bn_partial(
    const float* __restrict__ y, float* __restrict__ ss, int C, int rowsPer)
{
  const int c = threadIdx.x;
  const long long r0 = (long long)blockIdx.x * rowsPer;
  float s = 0.f, s2 = 0.f;
  for (int r = 0; r < rowsPer; ++r) {
    const float v = y[(r0 + r) * C + c];
    s += v; s2 += v * v;
  }
  atomicAdd(&ss[c], s);
  atomicAdd(&ss[C + c], s2);
}

__global__ void bn_finalize(float* __restrict__ ss, const float* __restrict__ g,
                            const float* __restrict__ beta, int C, float invCnt)
{
  const int c = threadIdx.x;
  const float mean = ss[c] * invCnt;
  const float var  = ss[C + c] * invCnt - mean * mean;
  const float sc   = g[c] * rsqrtf(var + 1e-5f);
  ss[2 * C + c] = sc;
  ss[3 * C + c] = beta[c] - mean * sc;
}

// ---------------------------------------------------------------------------
// BN apply + LeakyReLU(0.5) + optional residual add + f32/bf16 shadow writes
// + optional strided store into d_out slice [B, 4C, N] for layer `layer`.
// y, resid, xout, actout are [B, N, C]. Template removes runtime branching.
// ---------------------------------------------------------------------------
template <bool RESID, bool STORE_OUT>
__global__ __launch_bounds__(256) void bn_apply_kernel(
    const float* __restrict__ y, const float* __restrict__ ss,
    const float* __restrict__ resid, float* __restrict__ xout,
    bf16* __restrict__ actout, float* __restrict__ outBase,
    int layer, int C, int N)
{
  const long long idx = (long long)blockIdx.x * 256 + threadIdx.x;
  const int c = (int)(idx & (long long)(C - 1));
  float v = y[idx] * ss[2 * C + c] + ss[3 * C + c];
  v = (v >= 0.f) ? v : 0.5f * v;
  if constexpr (RESID) v += resid[idx];
  xout[idx]   = v;
  actout[idx] = (bf16)v;
  if constexpr (STORE_OUT) {
    const long long n = (idx / C) % N;
    const long long b = idx / ((long long)N * C);
    outBase[(b * (4LL * C) + (long long)layer * C + c) * N + n] = v;
  }
}

// ---------------------------------------------------------------------------
// Host orchestration. Workspace usage ≈ 212 MB (N x N attention buffers are
// processed per batch to bound scratch).
// ---------------------------------------------------------------------------
extern "C" void kernel_launch(void* const* d_in, const int* in_sizes, int n_in,
                              void* d_out, int out_size, void* d_ws, size_t ws_size,
                              hipStream_t stream)
{
  (void)in_sizes; (void)n_in; (void)out_size; (void)ws_size;
  const int Bv = 16, C = 256, N = 2048;
  const long long NC  = (long long)N * C;      // 524288
  const long long BNC = (long long)Bv * NC;    // 8388608
  const long long NN  = (long long)N * N;      // 4194304

  const float* x_in   = (const float*)d_in[0];
  const float* conv_w = (const float*)d_in[1];
  const float* conv_b = (const float*)d_in[2];
  const float* bn_g   = (const float*)d_in[3];
  const float* bn_b   = (const float*)d_in[4];
  const float* qw     = (const float*)d_in[5];
  const float* qb     = (const float*)d_in[6];
  const float* kb     = (const float*)d_in[7];
  const float* vw     = (const float*)d_in[8];
  const float* vb     = (const float*)d_in[9];
  const float* tw     = (const float*)d_in[10];
  const float* tb     = (const float*)d_in[11];
  const float* bng    = (const float*)d_in[12];
  const float* bnb    = (const float*)d_in[13];
  float* out = (float*)d_out;

  char* wp = (char*)d_ws;
  auto carve = [&](size_t bytes) -> void* {
    void* r = (void*)wp;
    wp += (bytes + 255) & ~(size_t)255;
    return r;
  };

  float* x_nc  = (float*)carve(BNC * 4);       // master activation [B,N,C]
  float* big0  = (float*)carve(BNC * 4);       // GEMM f32 scratch  [B,N,C]
  float* xr    = (float*)carve(BNC * 4);       // attention output  [B,N,C]
  bf16*  act   = (bf16*)carve(BNC * 2);        // bf16 shadow of x_nc
  bf16*  qbf   = (bf16*)carve(BNC * 2);
  bf16*  kbf   = (bf16*)carve(BNC * 2);
  bf16*  vbf   = (bf16*)carve(BNC * 2);        // v as [B,C,N] bf16
  bf16*  xmr   = (bf16*)carve(BNC * 2);        // bf16(x - x_r)
  float* E     = (float*)carve(NN * 4);        // one-batch attention matrix
  bf16*  attT  = (bf16*)carve(NN * 2);         // one-batch transposed att
  float* csum  = (float*)carve((size_t)N * 4);
  float* ss    = (float*)carve((size_t)4 * C * 4);
  bf16*  cw_bf = (bf16*)carve((size_t)2 * C * C * 2);
  bf16*  qw_bf = (bf16*)carve((size_t)4 * C * C * 2);
  bf16*  vw_bf = (bf16*)carve((size_t)4 * C * C * 2);
  bf16*  tw_bf = (bf16*)carve((size_t)4 * C * C * 2);

  // --- weight conversion to bf16 (exact-grid sizes) ---
  cvt_f32_bf16<<<(2 * C * C) / 256, 256, 0, stream>>>(conv_w, cw_bf);
  cvt_f32_bf16<<<(4 * C * C) / 256, 256, 0, stream>>>(qw, qw_bf);
  cvt_f32_bf16<<<(4 * C * C) / 256, 256, 0, stream>>>(vw, vw_bf);
  cvt_f32_bf16<<<(4 * C * C) / 256, 256, 0, stream>>>(tw, tw_bf);

  // --- input x [B,C,N] -> x_nc [B,N,C] (+ bf16 shadow) ---
  transpose_f32<<<dim3(N / 32, C / 32, Bv), dim3(32, 8), 0, stream>>>(
      x_in, x_nc, act, C, N, NC, NC);

  const float invCnt = 1.0f / (float)(Bv * N);

  // --- two front conv -> BN -> LeakyReLU blocks ---
  for (int i = 0; i < 2; ++i) {
    gemm_bf16_wmma<2, true, false><<<dim3(C / 128, N / 128, Bv), 256, 0, stream>>>(
        act, cw_bf + (long long)i * C * C, big0, (bf16*)nullptr,
        conv_b + i * C, /*K=*/C, /*ldc=*/C, NC, 0LL, NC);
    hipMemsetAsync(ss, 0, (size_t)4 * C * sizeof(float), stream);
    bn_partial<<<(Bv * N) / 64, 256, 0, stream>>>(big0, ss, C, 64);
    bn_finalize<<<1, C, 0, stream>>>(ss, bn_g + i * C, bn_b + i * C, C, invCnt);
    bn_apply_kernel<false, false><<<(int)(BNC / 256), 256, 0, stream>>>(
        big0, ss, (const float*)nullptr, x_nc, act, (float*)nullptr, 0, C, N);
  }

  // --- 4 self-attention layers ---
  for (int l = 0; l < 4; ++l) {
    const long long wl = (long long)l * C * C;

    // shared q/k GEMM: qk[b,n,o] = sum_c x[b,n,c] * qw[o,c]
    gemm_bf16_wmma<0, true, false><<<dim3(C / 128, N / 128, Bv), 256, 0, stream>>>(
        act, qw_bf + wl, big0, (bf16*)nullptr,
        (const float*)nullptr, C, C, NC, 0LL, NC);
    qk_bias_cvt<<<(int)(BNC / 256), 256, 0, stream>>>(
        big0, qb + l * C, kb + l * C, qbf, kbf, C);

    // v[b,c,n] = sum_ci vw[c,ci] * x[b,n,ci] + vb[c]  (bf16 out, [B,C,N])
    gemm_bf16_wmma<1, false, true><<<dim3(N / 128, C / 128, Bv), 256, 0, stream>>>(
        vw_bf + wl, act, (float*)nullptr, vbf,
        vb + l * C, /*K=*/C, /*ldc=*/N, 0LL, NC, (long long)C * N);

    // attention, batch-by-batch to bound workspace
    for (int b = 0; b < Bv; ++b) {
      const bf16* qp = qbf + (long long)b * NC;
      const bf16* kp = kbf + (long long)b * NC;
      // energy[n,m] = sum_c q[n,c] k[m,c]
      gemm_bf16_wmma<0, true, false><<<dim3(N / 128, N / 128, 1), 256, 0, stream>>>(
          qp, kp, E, (bf16*)nullptr, (const float*)nullptr, C, N, 0LL, 0LL, 0LL);
      softmax_rows<<<N, 256, 0, stream>>>(E, N);
      colsum_kernel<<<N / 256, 256, 0, stream>>>(E, csum, N);
      renorm_transpose<<<dim3(N / 32, N / 32), dim3(32, 8), 0, stream>>>(
          E, csum, attT, N);
      // x_r^T[m,c] = sum_n attT[m,n] * v[c,n]
      gemm_bf16_wmma<0, true, false><<<dim3(C / 128, N / 128, 1), 256, 0, stream>>>(
          attT, vbf + (long long)b * C * N, xr + (long long)b * NC, (bf16*)nullptr,
          (const float*)nullptr, /*K=*/N, /*ldc=*/C, 0LL, 0LL, 0LL);
    }

    // t = tw @ (x - x_r) + tb, then BN -> lrelu -> residual -> out slice l
    sub_cvt<<<(int)(BNC / 256), 256, 0, stream>>>(x_nc, xr, xmr);
    gemm_bf16_wmma<2, true, false><<<dim3(C / 128, N / 128, Bv), 256, 0, stream>>>(
        xmr, tw_bf + wl, big0, (bf16*)nullptr,
        tb + l * C, C, C, NC, 0LL, NC);
    hipMemsetAsync(ss, 0, (size_t)4 * C * sizeof(float), stream);
    bn_partial<<<(Bv * N) / 64, 256, 0, stream>>>(big0, ss, C, 64);
    bn_finalize<<<1, C, 0, stream>>>(ss, bng + l * C, bnb + l * C, C, invCnt);
    bn_apply_kernel<true, true><<<(int)(BNC / 256), 256, 0, stream>>>(
        big0, ss, x_nc, x_nc, act, out, l, C, N);
  }
}
